// DiffeqExactTraceAttention_51178830299532
// MI455X (gfx1250) — matrix operand order repaired and checked by reading, not compiled, not run on hardware
//
#include <hip/hip_runtime.h>

typedef _Float16 v8h  __attribute__((ext_vector_type(8)));
typedef _Float16 v16h __attribute__((ext_vector_type(16)));
typedef float    v8f  __attribute__((ext_vector_type(8)));

// ---------------------------------------------------------------------------
// WMMA helper: D = A(16x32 f16) * B(32x16 f16) + C(f32)
// ---------------------------------------------------------------------------
__device__ __forceinline__ v8f wmma_f16(v16h a, v16h b, v8f c) {
    return __builtin_amdgcn_wmma_f32_16x16x32_f16(false, a, false, b, (short)0, c,
                                                  false, false);
}

// A fragment (16x32, f16): lane holds row M=lane%16, K = (lane/16)*8 + {0..7, 16..23}
__device__ __forceinline__ v16h frag_a(const _Float16* p, int lda, int l16, int lh) {
    const _Float16* q = p + (size_t)l16 * lda + lh * 8;
    v8h lo = *(const v8h*)q;
    v8h hi = *(const v8h*)(q + 16);
    v16h r;
#pragma unroll
    for (int e = 0; e < 8; ++e) { r[e] = lo[e]; r[e + 8] = hi[e]; }
    return r;
}

// B fragment (32x16, f16) from Bt[N][K] row-major: lane holds col N=lane%16,
// K = (lane/16)*8 + {0..7, 16..23}
__device__ __forceinline__ v16h frag_b(const _Float16* p, int ldb, int l16, int lh) {
    const _Float16* q = p + (size_t)l16 * ldb + lh * 8;
    v8h lo = *(const v8h*)q;
    v8h hi = *(const v8h*)(q + 16);
    v16h r;
#pragma unroll
    for (int e = 0; e < 8; ++e) { r[e] = lo[e]; r[e + 8] = hi[e]; }
    return r;
}

__device__ __forceinline__ float redmax16(float v) {
#pragma unroll
    for (int m = 1; m < 16; m <<= 1) v = fmaxf(v, __shfl_xor(v, m, 32));
    return v;
}
__device__ __forceinline__ float redsum16(float v) {
#pragma unroll
    for (int m = 1; m < 16; m <<= 1) v += __shfl_xor(v, m, 32);
    return v;
}

// ---------------------------------------------------------------------------
// Weight prep: Out[n*Kp+k] = (k<Ksrc && mask(k,n)) ? W[k*N+n] : 0   (f32->f16)
// mode 0: none, 1: MADE m0, 2: MADE m1, 3: MADE m2   (D-1 == 15)
// ---------------------------------------------------------------------------
__global__ __launch_bounds__(256) void prep_wt_kernel(const float* __restrict__ W,
                                                      _Float16* __restrict__ Out,
                                                      int N, int Kp, int Ksrc, int mode) {
    int idx = blockIdx.x * 256 + threadIdx.x;
    if (idx >= N * Kp) return;
    int n = idx / Kp, k = idx % Kp;
    float v = 0.f;
    if (k < Ksrc) {
        bool m = true;
        if (mode == 1)      m = (n % 15) >= k;          // h_deg[n] >= in_deg[k]
        else if (mode == 2) m = (n % 15) >= (k % 15);   // h_deg[n] >= h_deg[k]
        else if (mode == 3) m = (n / 128) > (k % 15);   // out_deg[n] > h_deg[k]
        if (m) v = W[(size_t)k * N + n];
    }
    Out[idx] = (_Float16)v;
}

// x [2048,16] f32 -> xpad [2048,32] f16 (zero padded K)
__global__ __launch_bounds__(256) void prep_xpad_kernel(const float* __restrict__ x,
                                                        _Float16* __restrict__ xpad) {
    int idx = blockIdx.x * 256 + threadIdx.x;   // 2048*32
    int row = idx >> 5, c = idx & 31;
    float v = (c < 16) ? x[row * 16 + c] : 0.f;
    xpad[idx] = (_Float16)v;
}

// dimwise input rows: col0 = t, col1 = x_d, cols 66..95 = 0 (cols 2..65 = hfeat GEMM)
__global__ __launch_bounds__(256) void prep_zc_kernel(const float* __restrict__ t,
                                                      const float* __restrict__ x,
                                                      _Float16* __restrict__ zc, int cs) {
    int i = blockIdx.x * 256 + threadIdx.x;
    if (i >= 8192) return;
    _Float16* row = zc + (size_t)i * 96;
    row[0] = (_Float16)t[0];
    row[1] = (_Float16)x[cs + i];
#pragma unroll
    for (int c = 66; c < 96; ++c) row[c] = (_Float16)0.f;
}

// ---------------------------------------------------------------------------
// Generic WMMA GEMM with fused activation + tangent (JVP) epilogue.
// HASDOT/ACT are compile-time so the K-loop is branch-free straight-line WMMA.
//   Out    = act(A @ Bt^T + bias)                     (f16, row-major, ldo, +ocol)
//   OutDot = (1-Out^2) * (tseed ? tseed[col] : Adot@Bt^T)   (tanh layers only)
// block = 256 thr (8 waves); block tile 128(M) x 64(N); grid = (N/64, M/128)
// ---------------------------------------------------------------------------
template <bool HASDOT, int ACT>
__global__ __launch_bounds__(256) void gemm_kernel(
    const _Float16* __restrict__ A, const _Float16* __restrict__ Adot, int lda,
    const _Float16* __restrict__ Bt, int ldb,
    const float* __restrict__ bias, const float* __restrict__ tseed,
    _Float16* __restrict__ Out, _Float16* __restrict__ OutDot, int ldo, int ocol,
    int K) {
    const int w = threadIdx.x >> 5, lane = threadIdx.x & 31;
    const int l16 = lane & 15, lh = lane >> 4;
    const int row0 = blockIdx.y * 128 + w * 16;
    const int col0 = blockIdx.x * 64;
    v8f acc[4], accd[4];
#pragma unroll
    for (int j = 0; j < 4; ++j)
#pragma unroll
        for (int r = 0; r < 8; ++r) { acc[j][r] = 0.f; accd[j][r] = 0.f; }

#pragma unroll 2
    for (int k = 0; k < K; k += 32) {
        v16h a = frag_a(A + (size_t)row0 * lda + k, lda, l16, lh);
        v16h ad;
        if (HASDOT) ad = frag_a(Adot + (size_t)row0 * lda + k, lda, l16, lh);
#pragma unroll
        for (int j = 0; j < 4; ++j) {
            v16h bf = frag_b(Bt + (size_t)(col0 + j * 16) * ldb + k, ldb, l16, lh);
            acc[j] = wmma_f16(a, bf, acc[j]);
            if (HASDOT) accd[j] = wmma_f16(ad, bf, accd[j]);
        }
    }
#pragma unroll
    for (int j = 0; j < 4; ++j) {
        const int col = col0 + j * 16 + l16;
        const float bv = bias ? bias[col] : 0.f;
        const float ts = tseed ? tseed[col] : 0.f;
#pragma unroll
        for (int r = 0; r < 8; ++r) {
            const int row = row0 + lh * 8 + r;
            float v = acc[j][r] + bv;
            float hv, g = 0.f;
            if (ACT == 1)      hv = fmaxf(v, 0.f);
            else if (ACT == 2) { hv = tanhf(v); g = 1.f - hv * hv; }
            else               hv = v;
            Out[(size_t)row * ldo + ocol + col] = (_Float16)hv;
            if (OutDot) {
                float dv = tseed ? g * ts : g * accd[j][r];
                OutDot[(size_t)row * ldo + col] = (_Float16)dv;
            }
        }
    }
}

// ---------------------------------------------------------------------------
// Attention: one block per (b,d,h). 512 thr = 16 waves; wave w owns rows
// [16w,16w+16). S = q kT / sqrt(32), diag=-inf, softmax, O = P V / rowsum.
// ---------------------------------------------------------------------------
__global__ __launch_bounds__(512) void attn_kernel(const _Float16* __restrict__ qbuf,
                                                   const _Float16* __restrict__ kbuf,
                                                   const _Float16* __restrict__ vbuf,
                                                   _Float16* __restrict__ obuf) {
    __shared__ _Float16 kS[256 * 32];   // k slice, row-major [m][32] (== Bt for S)
    __shared__ _Float16 vtS[32 * 256];  // V^T [c][m]                 (== Bt for O)
    __shared__ _Float16 pS[16 * 512];   // per-wave P staging 16x32

    const int bid = blockIdx.x;
    const int b = bid >> 6, d = (bid >> 2) & 15, h = bid & 3;
    const int tid = threadIdx.x;
    {
        int m = tid >> 1, c0 = (tid & 1) << 4;
        const v8h* src = (const v8h*)(kbuf + ((size_t)(b * 256 + m)) * 128 + h * 32 + c0);
        v8h* dst = (v8h*)(kS + m * 32 + c0);
        dst[0] = src[0]; dst[1] = src[1];
    }
    for (int i = tid; i < 256 * 32; i += 512) {
        int m = i >> 5, c = i & 31;
        vtS[c * 256 + m] = vbuf[((size_t)(b * 256 + m)) * 128 + h * 32 + c];
    }
    __syncthreads();

    const int w = tid >> 5, lane = tid & 31;
    const int l16 = lane & 15, lh = lane >> 4;
    const int rowbase = w << 4;

    v16h a;
    {   // q rows for this wave (A fragment straight from global)
        const _Float16* qp =
            qbuf + ((size_t)(b * 256 + rowbase + l16)) * 2048 + d * 128 + h * 32 + lh * 8;
        v8h lo = *(const v8h*)qp;
        v8h hi = *(const v8h*)(qp + 16);
#pragma unroll
        for (int e = 0; e < 8; ++e) { a[e] = lo[e]; a[e + 8] = hi[e]; }
    }
    v8f s[16];
#pragma unroll
    for (int j = 0; j < 16; ++j) {
        const _Float16* kp = kS + (j * 16 + l16) * 32 + lh * 8;
        v8h lo = *(const v8h*)kp; v8h hi = *(const v8h*)(kp + 16);
        v16h bf;
#pragma unroll
        for (int e = 0; e < 8; ++e) { bf[e] = lo[e]; bf[e + 8] = hi[e]; }
        v8f z;
#pragma unroll
        for (int r = 0; r < 8; ++r) z[r] = 0.f;
        s[j] = wmma_f16(a, bf, z);
    }
    const float scale = 0.17677669529663687f;  // 1/sqrt(32)
    float lrow[8];
#pragma unroll
    for (int r = 0; r < 8; ++r) {
        const int grow = rowbase + lh * 8 + r;
        const int jd = grow >> 4;               // the single tile that holds the diagonal
        const bool hit = (grow & 15) == l16;    // this lane's column == diagonal column
        float mx = -3.0e38f;
#pragma unroll
        for (int j = 0; j < 16; ++j) {
            float val = s[j][r] * scale;
            if (hit && j == jd) val = -3.0e38f; // hollow diagonal
            s[j][r] = val;
            mx = fmaxf(mx, val);
        }
        mx = redmax16(mx);
        float sum = 0.f;
#pragma unroll
        for (int j = 0; j < 16; ++j) {
            float p = __expf(s[j][r] - mx);
            s[j][r] = p; sum += p;
        }
        lrow[r] = redsum16(sum);
    }
    v8f o0, o1;
#pragma unroll
    for (int r = 0; r < 8; ++r) { o0[r] = 0.f; o1[r] = 0.f; }
    _Float16* pw = pS + w * 512;
    for (int st = 0; st < 8; ++st) {
#pragma unroll
        for (int jj = 0; jj < 2; ++jj)
#pragma unroll
            for (int r = 0; r < 8; ++r)
                pw[(lh * 8 + r) * 32 + jj * 16 + l16] = (_Float16)s[st * 2 + jj][r];
        __builtin_amdgcn_wave_barrier();   // keep ds writes before reads
        v16h ap;
        {
            const _Float16* pp = pw + l16 * 32 + lh * 8;
            v8h lo = *(const v8h*)pp; v8h hi = *(const v8h*)(pp + 16);
#pragma unroll
            for (int e = 0; e < 8; ++e) { ap[e] = lo[e]; ap[e + 8] = hi[e]; }
        }
        {
            const _Float16* vp = vtS + l16 * 256 + st * 32 + lh * 8;
            v8h lo = *(const v8h*)vp; v8h hi = *(const v8h*)(vp + 16);
            v16h bf;
#pragma unroll
            for (int e = 0; e < 8; ++e) { bf[e] = lo[e]; bf[e + 8] = hi[e]; }
            o0 = wmma_f16(ap, bf, o0);
        }
        {
            const _Float16* vp = vtS + (16 + l16) * 256 + st * 32 + lh * 8;
            v8h lo = *(const v8h*)vp; v8h hi = *(const v8h*)(vp + 16);
            v16h bf;
#pragma unroll
            for (int e = 0; e < 8; ++e) { bf[e] = lo[e]; bf[e + 8] = hi[e]; }
            o1 = wmma_f16(ap, bf, o1);
        }
        __builtin_amdgcn_wave_barrier();
    }
#pragma unroll
    for (int tcol = 0; tcol < 2; ++tcol)
#pragma unroll
        for (int r = 0; r < 8; ++r) {
            int grow = rowbase + lh * 8 + r;
            float val = (tcol ? o1[r] : o0[r]) / lrow[r];
            size_t tokp = (size_t)(b * 256 + grow) * 16 + d;   // [B,N,D] token
            obuf[tokp * 128 + h * 32 + tcol * 16 + l16] = (_Float16)val;
        }
}

// final layer: y = h3 . w3 + b3 ; jac = h3dot . w3
__global__ __launch_bounds__(256) void final_kernel(const _Float16* __restrict__ h3,
                                                    const _Float16* __restrict__ h3d,
                                                    const float* __restrict__ w3,
                                                    const float* __restrict__ b3,
                                                    float* __restrict__ y,
                                                    float* __restrict__ jac, int M) {
    int i = blockIdx.x * 256 + threadIdx.x;
    if (i >= M) return;
    const _Float16* p  = h3  + (size_t)i * 128;
    const _Float16* pd = h3d + (size_t)i * 128;
    float s = 0.f, sd = 0.f;
#pragma unroll 4
    for (int c = 0; c < 128; ++c) {
        float wv = w3[c];
        s  += (float)p[c]  * wv;
        sd += (float)pd[c] * wv;
    }
    y[i]   = s + b3[0];
    jac[i] = sd;
}

// ---------------------------------------------------------------------------
extern "C" void kernel_launch(void* const* d_in, const int* in_sizes, int n_in,
                              void* d_out, int out_size, void* d_ws, size_t ws_size,
                              hipStream_t stream) {
    (void)in_sizes; (void)n_in; (void)out_size; (void)ws_size;
    const float* t    = (const float*)d_in[0];
    const float* x    = (const float*)d_in[1];
    const float* q_w0 = (const float*)d_in[2];  const float* q_b0 = (const float*)d_in[3];
    const float* k_w0 = (const float*)d_in[4];  const float* k_b0 = (const float*)d_in[5];
    const float* v_w0 = (const float*)d_in[6];  const float* v_b0 = (const float*)d_in[7];
    const float* q_w1 = (const float*)d_in[8];  const float* q_b1 = (const float*)d_in[9];
    const float* k_w1 = (const float*)d_in[10]; const float* k_b1 = (const float*)d_in[11];
    const float* v_w1 = (const float*)d_in[12]; const float* v_b1 = (const float*)d_in[13];
    const float* q_w2 = (const float*)d_in[14]; const float* q_b2 = (const float*)d_in[15];
    const float* k_w2 = (const float*)d_in[16]; const float* k_b2 = (const float*)d_in[17];
    const float* v_w2 = (const float*)d_in[18]; const float* v_b2 = (const float*)d_in[19];
    const float* p_w  = (const float*)d_in[20]; const float* p_b  = (const float*)d_in[21];
    const float* d_w0 = (const float*)d_in[22]; const float* d_b0 = (const float*)d_in[23];
    const float* d_w1 = (const float*)d_in[24]; const float* d_b1 = (const float*)d_in[25];
    const float* d_w2 = (const float*)d_in[26]; const float* d_b2 = (const float*)d_in[27];
    const float* d_w3 = (const float*)d_in[28]; const float* d_b3 = (const float*)d_in[29];
    float* out = (float*)d_out;
    char* ws = (char*)d_ws;

    size_t off = 0;
    auto take = [&](size_t bytes) { size_t r = off; off += (bytes + 255) & ~(size_t)255; return r; };
    // persistent weight region (~2 MB)
    size_t o_xpad = take(2048 * 32 * 2);
    size_t o_qw0 = take(256 * 32 * 2),  o_kw0 = take(256 * 32 * 2),  o_vw0 = take(256 * 32 * 2);
    size_t o_qw1 = take(256 * 256 * 2), o_kw1 = take(256 * 256 * 2), o_vw1 = take(256 * 256 * 2);
    size_t o_qw2 = take((size_t)2048 * 256 * 2);
    size_t o_kw2 = take(128 * 256 * 2), o_vw2 = take(128 * 256 * 2);
    size_t o_pw  = take(64 * 128 * 2);
    size_t o_dw0 = take(256 * 96 * 2);
    size_t o_dw1 = take(256 * 256 * 2);
    size_t o_dw2 = take(128 * 256 * 2);
    // region R1 (reused by dimwise chunk buffers after attention)
    size_t R1 = off;
    size_t o_hq1 = take((size_t)2048 * 256 * 2), o_hk1 = take((size_t)2048 * 256 * 2);
    size_t o_hv1 = take((size_t)2048 * 256 * 2), o_hq2 = take((size_t)2048 * 256 * 2);
    size_t o_hk2 = take((size_t)2048 * 256 * 2), o_hv2 = take((size_t)2048 * 256 * 2);
    size_t o_q = take((size_t)2048 * 2048 * 2);
    size_t o_k = take((size_t)2048 * 128 * 2);
    size_t o_v = take((size_t)2048 * 128 * 2);
    size_t o_obuf = take((size_t)32768 * 128 * 2);
    // dimwise chunk buffers: zc+h1+h1d+h2 alias R1 (13.5 MB <= 15.7 MB), rest fresh
    size_t o_zc  = R1;
    size_t o_h1  = o_zc  + (size_t)8192 * 96 * 2;
    size_t o_h1d = o_h1  + (size_t)8192 * 256 * 2;
    size_t o_h2  = o_h1d + (size_t)8192 * 256 * 2;
    size_t o_h2d = take((size_t)8192 * 256 * 2);
    size_t o_h3  = take((size_t)8192 * 128 * 2);
    size_t o_h3d = take((size_t)8192 * 128 * 2);

    auto F = [&](size_t o) { return (_Float16*)(ws + o); };

    // ---- weight prep (masks computed procedurally: m0/m1/m2 are index functions)
    prep_wt_kernel<<<32,   256, 0, stream>>>(q_w0, F(o_qw0), 256, 32, 16, 1);
    prep_wt_kernel<<<32,   256, 0, stream>>>(k_w0, F(o_kw0), 256, 32, 16, 0);
    prep_wt_kernel<<<32,   256, 0, stream>>>(v_w0, F(o_vw0), 256, 32, 16, 0);
    prep_wt_kernel<<<256,  256, 0, stream>>>(q_w1, F(o_qw1), 256, 256, 256, 2);
    prep_wt_kernel<<<256,  256, 0, stream>>>(k_w1, F(o_kw1), 256, 256, 256, 0);
    prep_wt_kernel<<<256,  256, 0, stream>>>(v_w1, F(o_vw1), 256, 256, 256, 0);
    prep_wt_kernel<<<2048, 256, 0, stream>>>(q_w2, F(o_qw2), 2048, 256, 256, 3);
    prep_wt_kernel<<<128,  256, 0, stream>>>(k_w2, F(o_kw2), 128, 256, 256, 0);
    prep_wt_kernel<<<128,  256, 0, stream>>>(v_w2, F(o_vw2), 128, 256, 256, 0);
    prep_wt_kernel<<<32,   256, 0, stream>>>(p_w,  F(o_pw),  64, 128, 128, 0);
    prep_wt_kernel<<<96,   256, 0, stream>>>(d_w0, F(o_dw0), 256, 96, 66, 0);
    prep_wt_kernel<<<256,  256, 0, stream>>>(d_w1, F(o_dw1), 256, 256, 256, 0);
    prep_wt_kernel<<<128,  256, 0, stream>>>(d_w2, F(o_dw2), 128, 256, 256, 0);
    prep_xpad_kernel<<<256, 256, 0, stream>>>(x, F(o_xpad));

    // ---- q/k/v token MLPs (WMMA GEMM chain over 2048 tokens)
    gemm_kernel<false, 1><<<dim3(4, 16),  256, 0, stream>>>(F(o_xpad), nullptr, 32,  F(o_qw0), 32,  q_b0, nullptr, F(o_hq1), nullptr, 256, 0, 32);
    gemm_kernel<false, 2><<<dim3(4, 16),  256, 0, stream>>>(F(o_xpad), nullptr, 32,  F(o_kw0), 32,  k_b0, nullptr, F(o_hk1), nullptr, 256, 0, 32);
    gemm_kernel<false, 2><<<dim3(4, 16),  256, 0, stream>>>(F(o_xpad), nullptr, 32,  F(o_vw0), 32,  v_b0, nullptr, F(o_hv1), nullptr, 256, 0, 32);
    gemm_kernel<false, 1><<<dim3(4, 16),  256, 0, stream>>>(F(o_hq1),  nullptr, 256, F(o_qw1), 256, q_b1, nullptr, F(o_hq2), nullptr, 256, 0, 256);
    gemm_kernel<false, 2><<<dim3(4, 16),  256, 0, stream>>>(F(o_hk1),  nullptr, 256, F(o_kw1), 256, k_b1, nullptr, F(o_hk2), nullptr, 256, 0, 256);
    gemm_kernel<false, 2><<<dim3(4, 16),  256, 0, stream>>>(F(o_hv1),  nullptr, 256, F(o_vw1), 256, v_b1, nullptr, F(o_hv2), nullptr, 256, 0, 256);
    gemm_kernel<false, 0><<<dim3(32, 16), 256, 0, stream>>>(F(o_hq2),  nullptr, 256, F(o_qw2), 256, q_b2, nullptr, F(o_q),   nullptr, 2048, 0, 256);
    gemm_kernel<false, 0><<<dim3(2, 16),  256, 0, stream>>>(F(o_hk2),  nullptr, 256, F(o_kw2), 256, k_b2, nullptr, F(o_k),   nullptr, 128, 0, 256);
    gemm_kernel<false, 0><<<dim3(2, 16),  256, 0, stream>>>(F(o_hv2),  nullptr, 256, F(o_vw2), 256, v_b2, nullptr, F(o_v),   nullptr, 128, 0, 256);

    // ---- attention: 512 blocks = B*D*NH, writes o as [B,N,D,H] f16
    attn_kernel<<<512, 512, 0, stream>>>(F(o_q), F(o_k), F(o_v), F(o_obuf));

    // ---- dimwise net + exact JVP, 4 chunks of 8192 rows
    for (int ci = 0; ci < 4; ++ci) {
        const int cs = ci * 8192;
        prep_zc_kernel<<<32, 256, 0, stream>>>(t, x, F(o_zc), cs);
        // hfeat = o @ p_w + p_b -> z cols [2..66)
        gemm_kernel<false, 0><<<dim3(1, 64), 256, 0, stream>>>(F(o_obuf) + (size_t)cs * 128, nullptr, 128,
                                                               F(o_pw), 128, p_b, nullptr,
                                                               F(o_zc), nullptr, 96, 2, 128);
        // L1: tangent seed = W0 row 1 (d/dx of the x channel)
        gemm_kernel<false, 2><<<dim3(4, 64), 256, 0, stream>>>(F(o_zc), nullptr, 96, F(o_dw0), 96,
                                                               d_b0, d_w0 + 256,
                                                               F(o_h1), F(o_h1d), 256, 0, 96);
        // L2 / L3: primal + tangent fused (shared B fragments)
        gemm_kernel<true, 2><<<dim3(4, 64), 256, 0, stream>>>(F(o_h1), F(o_h1d), 256, F(o_dw1), 256,
                                                              d_b1, nullptr,
                                                              F(o_h2), F(o_h2d), 256, 0, 256);
        gemm_kernel<true, 2><<<dim3(2, 64), 256, 0, stream>>>(F(o_h2), F(o_h2d), 256, F(o_dw2), 256,
                                                              d_b2, nullptr,
                                                              F(o_h3), F(o_h3d), 128, 0, 256);
        final_kernel<<<32, 256, 0, stream>>>(F(o_h3), F(o_h3d), d_w3, d_b3,
                                             out + cs, out + 32768 + cs, 8192);
    }
}
